// MoEModel_87557203297090
// MI455X (gfx1250) — matrix operand order repaired and checked by reading, not compiled, not run on hardware
//
#include <hip/hip_runtime.h>
#include <cfloat>
#include <cmath>

#define DI __device__ __forceinline__

typedef __attribute__((ext_vector_type(16))) __bf16 v16bf;
typedef __attribute__((ext_vector_type(8)))  __bf16 v8bf;
typedef __attribute__((ext_vector_type(8)))  float  v8f;

constexpr int Bn = 2048, Sn = 4, En = 64, Hn = 985, Kn = 4, HP = 1024;

// ---- output offsets (float elements, concatenated in reference return order) ----
constexpr size_t O_OUTSEL = 0;        // (B,4,1)
constexpr size_t O_EMO    = 8192;     // (B,1)
constexpr size_t O_TKI    = 10240;    // (B,4) indices as float
constexpr size_t O_SELIDX = 18432;    // (B,1) indices as float
constexpr size_t O_SIM    = 20480;    // (B,4)
constexpr size_t O_GATES  = 28672;    // (B,E)
constexpr size_t O_YVIEW  = 159744;   // (B,E,1)
constexpr size_t O_SELP   = 290816;   // (B,E)
constexpr size_t O_LAST   = 421888;   // (B,)
constexpr size_t O_TKG    = 423936;   // (B,4)

// ---- workspace offsets (bytes) ----
constexpr size_t W_X0BF = 0;                           // B*HP bf16
constexpr size_t W_EMT  = W_X0BF + (size_t)Bn*HP*2;    // HP*HP bf16 (expert_min^T, padded)
constexpr size_t W_GW   = W_EMT  + (size_t)HP*HP*2;    // E*HP bf16 (gate_weights, padded)
constexpr size_t W_W2   = W_GW   + (size_t)En*HP*2;    // E*HP bf16 (W @ out_w, padded)
constexpr size_t W_EMS  = W_W2   + (size_t)En*HP*2;    // B*HP f32
constexpr size_t W_RP   = W_EMS  + (size_t)Bn*HP*4;    // B*E f32 (random_p)
constexpr size_t W_AO   = W_RP   + (size_t)Bn*En*4;    // B*K f32 (out_w dot)
constexpr size_t W_AS   = W_AO   + (size_t)Bn*Kn*4;    // B*K f32 (ems dot)
constexpr size_t W_AN   = W_AS   + (size_t)Bn*Kn*4;    // B*K f32 (norm^2)
constexpr size_t W_TKI  = W_AN   + (size_t)Bn*Kn*4;    // B*K int
constexpr size_t W_RL   = W_TKI  + (size_t)Bn*Kn*4;    // B*K int (routed row list)
constexpr size_t W_CNT  = W_RL   + (size_t)Bn*Kn*4;    // E int
constexpr size_t W_OFF  = W_CNT  + 256;                // E+1 int
constexpr size_t W_CUR  = W_OFF  + 512;                // E int

// ---------------- wave reductions (wave32) ----------------
DI float red_half16(float v) {               // sum within each 16-lane half
  v += __shfl_xor(v, 1); v += __shfl_xor(v, 2);
  v += __shfl_xor(v, 4); v += __shfl_xor(v, 8);
  return v;
}
DI float red_full32(float v) {
  v = red_half16(v); v += __shfl_xor(v, 16);
  return v;
}

// ---------------- WMMA fragment loaders ----------------
// A (16x32 bf16, M x K): lane<16 -> row=lane, K k0+0..15
//                        lane>=16 -> row=lane-16, K k0+16..31
DI v16bf load_a_frag(const __bf16* Arow, int k0, int lane) {
  int off = (lane < 16) ? 0 : 16;
  const __bf16* p = Arow + k0 + off;
  v8bf lo = *(const v8bf*)p;
  v8bf hi = *(const v8bf*)(p + 8);
  v16bf f;
#pragma unroll
  for (int j = 0; j < 8; ++j) { f[j] = lo[j]; f[8 + j] = hi[j]; }
  return f;
}
// B (32x16 bf16, K x N) supplied as transposed weights Bt (N x K row-major):
// lane<16 -> col=lane, K k0+0..15 ; lane>=16 -> col=lane-16, K k0+16..31
DI v16bf load_b_frag(const __bf16* Bt, int ldb, int col, int k0, int lane) {
  int off = (lane < 16) ? 0 : 16;
  const __bf16* p = Bt + (size_t)col * ldb + k0 + off;
  v8bf lo = *(const v8bf*)p;
  v8bf hi = *(const v8bf*)(p + 8);
  v16bf f;
#pragma unroll
  for (int j = 0; j < 8; ++j) { f[j] = lo[j]; f[8 + j] = hi[j]; }
  return f;
}

// ---------------- threefry2x32 (JAX-compatible, key = (0,42)) ----------------
DI unsigned tf_rotl(unsigned x, int r) { return (x << r) | (x >> (32 - r)); }
DI void tf2x32(unsigned c0, unsigned c1, unsigned& o0, unsigned& o1) {
  const unsigned k0 = 0u, k1 = 42u, k2 = 0x1BD11BDAu ^ k0 ^ k1;
  unsigned x0 = c0 + k0, x1 = c1 + k1;
#define TF_R4(a,b,c,d) \
  x0 += x1; x1 = tf_rotl(x1,a); x1 ^= x0; \
  x0 += x1; x1 = tf_rotl(x1,b); x1 ^= x0; \
  x0 += x1; x1 = tf_rotl(x1,c); x1 ^= x0; \
  x0 += x1; x1 = tf_rotl(x1,d); x1 ^= x0;
  TF_R4(13,15,26,6)  x0 += k1; x1 += k2 + 1u;
  TF_R4(17,29,16,24) x0 += k2; x1 += k0 + 2u;
  TF_R4(13,15,26,6)  x0 += k0; x1 += k1 + 3u;
  TF_R4(17,29,16,24) x0 += k1; x1 += k2 + 4u;
  TF_R4(13,15,26,6)  x0 += k2; x1 += k0 + 5u;
#undef TF_R4
  o0 = x0; o1 = x1;
}
DI float bits_to_uniform(unsigned b) {
  return __uint_as_float((b >> 9) | 0x3f800000u) - 1.0f;
}

// ---------------- setup / conversion kernels ----------------
__global__ void k_zero_small(int* cnt, int* cur) {
  int t = threadIdx.x;
  if (t < En) { cnt[t] = 0; cur[t] = 0; }
}
__global__ void k_cvt_x0(const float* x, unsigned short* x0bf) {
  int idx = blockIdx.x * blockDim.x + threadIdx.x;   // B*HP
  int b = idx >> 10, h = idx & 1023;
  float v = (h < Hn) ? x[(size_t)b * Sn * Hn + h] : 0.f;   // x[:,0,:]
  ((__bf16*)x0bf)[idx] = (__bf16)v;
}
__global__ void k_cvt_gw(const float* gw, unsigned short* gwbf) {
  int idx = blockIdx.x * blockDim.x + threadIdx.x;   // E*HP
  int e = idx >> 10, h = idx & 1023;
  float v = (h < Hn) ? gw[(size_t)e * Hn + h] : 0.f;
  ((__bf16*)gwbf)[idx] = (__bf16)v;
}
__global__ void k_cvt_emt(const float* em, unsigned short* emt) {
  int idx = blockIdx.x * blockDim.x + threadIdx.x;   // HP*HP: [n][h] = em[h][n]
  int n = idx >> 10, h = idx & 1023;
  float v = (n < Hn && h < Hn) ? em[(size_t)h * Hn + n] : 0.f;
  ((__bf16*)emt)[idx] = (__bf16)v;
}
// W2[e,h] = sum_d expert_weights[e,h,d] * out_w[d]   (one wave per (e,h) row)
__global__ void k_w2(const float* W, const float* outw, unsigned short* w2bf) {
  int lane = threadIdx.x & 31, wid = threadIdx.x >> 5;
  int row = blockIdx.x * 8 + wid;                    // E*HP rows
  int e = row >> 10, h = row & 1023;
  float acc = 0.f;
  if (h < Hn) {
    const float* p = W + (size_t)e * Hn * Hn + (size_t)h * Hn;
    for (int d = lane; d < Hn; d += 32) acc += p[d] * outw[d];
  }
  acc = red_full32(acc);
  if (lane == 0) ((__bf16*)w2bf)[row] = (__bf16)((h < Hn) ? acc : 0.f);
}

// ---------------- generic bf16 WMMA GEMM: C[M,N] = A(MxHP) * Bt(NxHP)^T ----------------
__global__ void __launch_bounds__(32)
k_gemm_nt(const unsigned short* Abf, const unsigned short* Btbf,
          float* C, int ldc, int Nvalid, const float* bias) {
  int lane = threadIdx.x;
  int row0 = blockIdx.x * 16, col0 = blockIdx.y * 16;
  const __bf16* A  = (const __bf16*)Abf;
  const __bf16* Bt = (const __bf16*)Btbf;
  const __bf16* Arow = A + (size_t)(row0 + (lane & 15)) * HP;
  int col = col0 + (lane & 15);
  v8f c = {};
  for (int k0 = 0; k0 < HP; k0 += 32) {
    v16bf a = load_a_frag(Arow, k0, lane);
    v16bf b = load_b_frag(Bt, HP, col, k0, lane);
    c = __builtin_amdgcn_wmma_f32_16x16x32_bf16(false, a, false, b,
                                                (short)0, c, false, false);
  }
  float bb = bias ? bias[0] : 0.f;
  if (col < Nvalid) {
#pragma unroll
    for (int r = 0; r < 8; ++r) {
      int row = row0 + r + ((lane < 16) ? 0 : 8);
      C[(size_t)row * ldc + col] = c[r] + bb;
    }
  }
}

// ---------------- top-k, threefry pick, routing counts ----------------
__global__ void k_topk(const float* out_gates, float* out, float* rp,
                       int* tki, int* cnt) {
  int b = blockIdx.x * blockDim.x + threadIdx.x;
  if (b >= Bn) return;
  const float* g = out_gates + (size_t)b * En;
  // stable top-4 of gates
  int idx4[4]; float val4[4];
#pragma unroll
  for (int t = 0; t < 4; ++t) {
    float bv = -FLT_MAX; int bi = 0;
    for (int e = 0; e < En; ++e) {
      bool used = false;
      for (int u = 0; u < t; ++u) used |= (idx4[u] == e);
      if (used) continue;
      float v = g[e];
      if (v > bv) { bv = v; bi = e; }
    }
    idx4[t] = bi; val4[t] = bv;
  }
  // select_probability
  float* sp = out + O_SELP + (size_t)b * En;
  for (int e = 0; e < En; ++e) sp[e] = 0.f;
  for (int t = 0; t < 3; ++t) sp[idx4[t]] = val4[t];
  // random_p = 0.1*uniform + 0.9*gates  (threefry, key 42, counter b*64+e)
  float* rpr = rp + (size_t)b * En;
  for (int e = 0; e < En; ++e) {
    unsigned i = (unsigned)(b * En + e), o0, o1, bits;
    const unsigned half = (unsigned)(Bn * En / 2);
    if (i < half) { tf2x32(i, i + half, o0, o1); bits = o0; }
    else          { tf2x32(i - half, i, o0, o1); bits = o1; }
    float u = bits_to_uniform(bits);
    rpr[e] = 0.1f * u + 0.9f * g[e];
  }
  // stable top-4 of random_p
  int p4[4];
#pragma unroll
  for (int t = 0; t < 4; ++t) {
    float bv = -FLT_MAX; int bi = 0;
    for (int e = 0; e < En; ++e) {
      bool used = false;
      for (int u = 0; u < t; ++u) used |= (p4[u] == e);
      if (used) continue;
      float v = rpr[e];
      if (v > bv) { bv = v; bi = e; }
    }
    p4[t] = bi;
  }
  // first pick not among top-3 gates
  int first = 0;
  for (int k = 0; k < 4; ++k) {
    bool member = (p4[k] == idx4[0]) | (p4[k] == idx4[1]) | (p4[k] == idx4[2]);
    if (!member) { first = k; break; }
  }
  int newlast = p4[first];
  int tk[4] = { idx4[0], idx4[1], idx4[2], newlast };
  for (int k = 0; k < 4; ++k) {
    out[O_TKI + (size_t)b * 4 + k] = (float)tk[k];
    out[O_TKG + (size_t)b * 4 + k] = val4[k];
    tki[b * 4 + k] = tk[k];
    atomicAdd(&cnt[tk[k]], 1);
  }
  out[O_LAST + b] = (float)newlast;
}

__global__ void k_scan(const int* cnt, int* offs) {
  if (threadIdx.x == 0 && blockIdx.x == 0) {
    int acc = 0;
    for (int e = 0; e < En; ++e) { offs[e] = acc; acc += cnt[e]; }
    offs[En] = acc;
  }
}
__global__ void k_scatter(const int* tki, const int* offs, int* cur, int* rlist) {
  int t = blockIdx.x * blockDim.x + threadIdx.x;   // B*K
  if (t >= Bn * Kn) return;
  int e = tki[t];
  int pos = offs[e] + atomicAdd(&cur[e], 1);
  rlist[pos] = t;                                   // packed b*4+k
}

// ---------------- grouped expert GEMM with fused epilogue ----------------
// For each routed row (b,k) of expert e: emb = x0[b] @ W_e  (1 x H)
// Accumulate: dot(emb,out_w), dot(emb,ems[b]), ||emb||^2  -> aux arrays.
__global__ void __launch_bounds__(32)
k_grouped(const unsigned short* x0bf, const float* W, const float* outw,
          const float* ems, const int* rlist, const int* offs,
          float* auxO, float* auxS, float* auxN) {
  int e = blockIdx.x, seg = blockIdx.y, lane = threadIdx.x;
  int beg = offs[e], cnt = offs[e + 1] - beg;
  if (seg * 16 >= cnt) return;
  int rid = -1;
  if (lane < 16 && seg * 16 + lane < cnt) rid = rlist[beg + seg * 16 + lane];
  int bvec = (rid >= 0) ? (rid >> 2) : 0;           // b of row (lane) on lanes 0..15
  int myb  = __shfl(bvec, lane & 15);
  const __bf16* A  = (const __bf16*)x0bf;
  const __bf16* Arow = A + (size_t)myb * HP;
  const float*  We = W + (size_t)e * Hn * Hn;
  int col = (lane & 15);
  int koff = (lane < 16) ? 0 : 16;
  float accO = 0.f, accS = 0.f, accN = 0.f;
  for (int col0 = 0; col0 < HP; col0 += 16) {
    int d = col0 + col;                              // output feature index
    v8f c = {};
    for (int k0 = 0; k0 < HP; k0 += 32) {
      v16bf a = load_a_frag(Arow, k0, lane);
      v16bf bf;
#pragma unroll
      for (int j = 0; j < 16; ++j) {
        int kk = k0 + koff + j;
        float wv = (kk < Hn && d < Hn) ? We[(size_t)kk * Hn + d] : 0.f;
        bf[j] = (__bf16)wv;
      }
      c = __builtin_amdgcn_wmma_f32_16x16x32_bf16(false, a, false, bf,
                                                  (short)0, c, false, false);
    }
    float w = (d < Hn) ? outw[d] : 0.f;
#pragma unroll
    for (int r = 0; r < 8; ++r) {
      int row = r + ((lane < 16) ? 0 : 8);
      int rb = __shfl(bvec, row);
      float v = c[r];
      float ev = ems[(size_t)rb * HP + d];
      float po = red_half16(v * w);
      float ps = red_half16(v * ev);
      float pn = red_half16(v * v);
      if ((lane & 15) == r) { accO += po; accS += ps; accN += pn; }
    }
  }
  // lanes 0..7 hold rows 0..7, lanes 16..23 hold rows 8..15
  if ((lane & 15) < 8) {
    int rr = (lane < 16) ? lane : (lane - 8);
    int rid2 = __shfl(rid, rr & 15);
    // for high half, row = lane-8 lives on source lane (lane-8): recompute properly
    rid2 = __shfl(rid, (lane < 16) ? lane : (lane - 8));
    if (rid2 >= 0) { auxO[rid2] = accO; auxS[rid2] = accS; auxN[rid2] = accN; }
  }
}

// ---------------- finalize: norms, sims, select, expert_min_out ----------------
__global__ void k_finalize(const float* ems, const float* outw, const float* outb,
                           const float* auxO, const float* auxS, const float* auxN,
                           const int* tki, float* out) {
  int lane = threadIdx.x & 31, wid = threadIdx.x >> 5;
  int b = blockIdx.x * 8 + wid;
  const float* er = ems + (size_t)b * HP;
  float s2 = 0.f, so = 0.f;
  for (int h = lane; h < Hn; h += 32) { float v = er[h]; s2 += v * v; so += v * outw[h]; }
  s2 = red_full32(s2); so = red_full32(so);
  if (lane == 0) {
    float ob = outb[0];
    out[O_EMO + b] = so + ob;
    float n1 = fmaxf(sqrtf(s2), 1e-8f);
    float best = -FLT_MAX; int bestk = 0;
#pragma unroll
    for (int k = 0; k < 4; ++k) {
      int idx = b * 4 + k;
      float n2 = fmaxf(sqrtf(auxN[idx]), 1e-8f);
      float sim = auxS[idx] / (n1 * n2);
      out[O_SIM + idx] = sim;
      out[O_OUTSEL + idx] = auxO[idx] + ob;
      if (sim > best) { best = sim; bestk = k; }
    }
    out[O_SELIDX + b] = (float)tki[b * 4 + bestk];
  }
}

// ---------------- launch ----------------
extern "C" void kernel_launch(void* const* d_in, const int* in_sizes, int n_in,
                              void* d_out, int out_size, void* d_ws, size_t ws_size,
                              hipStream_t stream) {
  const float* x    = (const float*)d_in[0];
  const float* gw   = (const float*)d_in[1];
  const float* expW = (const float*)d_in[2];
  const float* emin = (const float*)d_in[3];
  const float* outw = (const float*)d_in[4];
  const float* outb = (const float*)d_in[5];
  float* out = (float*)d_out;
  char* ws = (char*)d_ws;

  unsigned short* x0bf = (unsigned short*)(ws + W_X0BF);
  unsigned short* emt  = (unsigned short*)(ws + W_EMT);
  unsigned short* gwbf = (unsigned short*)(ws + W_GW);
  unsigned short* w2bf = (unsigned short*)(ws + W_W2);
  float* ems = (float*)(ws + W_EMS);
  float* rp  = (float*)(ws + W_RP);
  float* auxO = (float*)(ws + W_AO);
  float* auxS = (float*)(ws + W_AS);
  float* auxN = (float*)(ws + W_AN);
  int* tki   = (int*)(ws + W_TKI);
  int* rlist = (int*)(ws + W_RL);
  int* cnt   = (int*)(ws + W_CNT);
  int* offs  = (int*)(ws + W_OFF);
  int* cur   = (int*)(ws + W_CUR);

  k_zero_small<<<1, 128, 0, stream>>>(cnt, cur);
  k_cvt_x0<<<(Bn * HP) / 256, 256, 0, stream>>>(x, x0bf);
  k_cvt_gw<<<(En * HP) / 256, 256, 0, stream>>>(gw, gwbf);
  k_cvt_emt<<<(HP * HP) / 256, 256, 0, stream>>>(emin, emt);
  k_w2<<<(En * HP) / 8, 256, 0, stream>>>(expW, outw, w2bf);

  // gates = x0 @ gate_weights^T
  k_gemm_nt<<<dim3(Bn / 16, En / 16), 32, 0, stream>>>(
      x0bf, gwbf, out + O_GATES, En, En, nullptr);
  // y_view = x0 @ W2^T + out_b
  k_gemm_nt<<<dim3(Bn / 16, En / 16), 32, 0, stream>>>(
      x0bf, w2bf, out + O_YVIEW, En, En, outb);
  // ems = x0 @ expert_min
  k_gemm_nt<<<dim3(Bn / 16, HP / 16), 32, 0, stream>>>(
      x0bf, emt, ems, HP, HP, nullptr);

  k_topk<<<Bn / 256, 256, 0, stream>>>(out + O_GATES, out, rp, tki, cnt);
  k_scan<<<1, 1, 0, stream>>>(cnt, offs);
  k_scatter<<<(Bn * Kn) / 256, 256, 0, stream>>>(tki, offs, cur, rlist);

  // grouped expert GEMM: max segments per expert = B/16 = 128
  k_grouped<<<dim3(En, Bn / 16), 32, 0, stream>>>(
      x0bf, expW, outw, ems, rlist, offs, auxO, auxS, auxN);

  k_finalize<<<Bn / 8, 256, 0, stream>>>(ems, outw, outb, auxO, auxS, auxN, tki, out);
}
